// Actor_42185168781388
// MI455X (gfx1250) — matrix operand order repaired and checked by reading, not compiled, not run on hardware
//
#include <hip/hip_runtime.h>
#include <hip/hip_bf16.h>

typedef __attribute__((ext_vector_type(16))) _Float16 v16h;
typedef __attribute__((ext_vector_type(8)))  _Float16 v8h;
typedef __attribute__((ext_vector_type(8)))  float    v8f;

#define WMMA_F16(a, b, c) \
  __builtin_amdgcn_wmma_f32_16x16x32_f16(false, (a), false, (b), (short)0, (c), false, false)

__device__ __forceinline__ v8f vzero8() {
  v8f z = {0.f, 0.f, 0.f, 0.f, 0.f, 0.f, 0.f, 0.f};
  return z;
}

// ---------------------------------------------------------------------------
// Zero a float region of workspace (accumulators must be zero every call).
// ---------------------------------------------------------------------------
__global__ void zero_kernel(float* __restrict__ p, long n) {
  long i = (long)blockIdx.x * blockDim.x + threadIdx.x;
  if (i < n) p[i] = 0.f;
}

// ---------------------------------------------------------------------------
// One-time f32 -> f16 conversion (x_nodes), so edge gathers are 16B f16 runs
// with no per-edge conversion in the hot WMMA loop.
// ---------------------------------------------------------------------------
__global__ void to_f16_kernel(const float* __restrict__ src, _Float16* __restrict__ dst,
                              long n) {
  long i = (long)blockIdx.x * blockDim.x + threadIdx.x;
  if (i < n) dst[i] = (_Float16)src[i];
}

// ---------------------------------------------------------------------------
// Pack an f32 row-major [K x Nn] weight matrix into f16 WMMA B-fragments.
// Fragment order: ((kt*NT + nt)*32 + lane)*16 + i ; per-lane: N = nt*16+(lane&15),
// K = kt*32 + (lane>=16 ? 8 : 0) + (i&7) + (i>=8 ? 16 : 0).  Zero-pads K/N beyond the real matrix.
// ---------------------------------------------------------------------------
__global__ void pack_w_kernel(const float* __restrict__ W, _Float16* __restrict__ dst,
                              int K, int Nn, int KT, int NT) {
  int idx = blockIdx.x * blockDim.x + threadIdx.x;
  int total = KT * NT * 512;
  if (idx >= total) return;
  int i    = idx & 15;
  int lane = (idx >> 4) & 31;
  int frag = idx >> 9;
  int nt = frag % NT;
  int kt = frag / NT;
  int n = nt * 16 + (lane & 15);
  int k = kt * 32 + ((lane >= 16) ? 8 : 0) + (i & 7) + ((i >= 8) ? 16 : 0);
  float v = (k < K && n < Nn) ? W[(size_t)k * Nn + n] : 0.f;
  dst[idx] = (_Float16)v;
}

// ---------------------------------------------------------------------------
// Edge MLP layer 1: per wave 16 edges. in = cat[x[dst], x[src]] (128 f16),
// relu(in @ Wa + ba) @ Wb + bb  -> atomic scatter-add into h_sum[dst], count++.
// ---------------------------------------------------------------------------
__global__ __launch_bounds__(256) void edge_mlp1_kernel(
    const _Float16* __restrict__ xn16, const int* __restrict__ ei, long E,
    const _Float16* __restrict__ pA, const _Float16* __restrict__ pB,
    const float* __restrict__ ba, const float* __restrict__ bb,
    float* __restrict__ hsum, float* __restrict__ cnt) {
  __shared__ __align__(32) _Float16 sStage[8][16 * 128];
  __shared__ int sDst[8][16];
  const int w    = threadIdx.x >> 5;
  const int lane = threadIdx.x & 31;
  const int m16  = lane & 15;
  const int hi   = lane >> 4;
  const long base = ((long)blockIdx.x * 8 + w) * 16;
  long e = base + m16;
  const bool valid = (e < E);
  const long ec = valid ? e : 0;
  const int src = ei[ec];
  const int dst = ei[E + ec];
  if (hi == 0) {
    sDst[w][m16] = valid ? dst : -1;
    if (valid) unsafeAtomicAdd(&cnt[dst], 1.0f);
  }
  const _Float16* rowD = xn16 + (long)dst * 64;
  const _Float16* rowS = xn16 + (long)src * 64;

  v8f acc[8];
#pragma unroll
  for (int nt = 0; nt < 8; ++nt) acc[nt] = vzero8();

  // GEMM1: [16x128] @ [128x128]
#pragma unroll
  for (int kt = 0; kt < 4; ++kt) {
    v16h a;
    const int f0 = kt * 32 + hi * 8;
    const int f1 = f0 + 16;
    const _Float16* p0 = (f0 < 64) ? (rowD + f0) : (rowS + (f0 - 64));
    const _Float16* p1 = (f1 < 64) ? (rowD + f1) : (rowS + (f1 - 64));
    v8h u0 = *(const v8h*)p0;
    v8h u1 = *(const v8h*)p1;
#pragma unroll
    for (int j = 0; j < 8; ++j) { a[j] = u0[j]; a[8 + j] = u1[j]; }
#pragma unroll
    for (int nt = 0; nt < 8; ++nt) {
      v16h b = *(const v16h*)(pA + (((kt * 8 + nt) * 32 + lane) << 4));
      acc[nt] = WMMA_F16(a, b, acc[nt]);
    }
  }
  // bias + relu -> per-wave LDS stage (row-major [16][128] f16)
  _Float16* st = sStage[w];
#pragma unroll
  for (int nt = 0; nt < 8; ++nt) {
    const int n = nt * 16 + m16;
    const float bv = ba[n];
#pragma unroll
    for (int r = 0; r < 8; ++r) {
      const int m = r + hi * 8;
      float v = acc[nt][r] + bv;
      st[m * 128 + n] = (_Float16)(v > 0.f ? v : 0.f);
    }
  }
  // GEMM2: [16x128] @ [128x128]
#pragma unroll
  for (int nt = 0; nt < 8; ++nt) acc[nt] = vzero8();
#pragma unroll
  for (int kt = 0; kt < 4; ++kt) {
    v16h a;
    const int f0 = kt * 32 + hi * 8;
    v8h u0 = *(const v8h*)(st + m16 * 128 + f0);
    v8h u1 = *(const v8h*)(st + m16 * 128 + f0 + 16);
#pragma unroll
    for (int j = 0; j < 8; ++j) { a[j] = u0[j]; a[8 + j] = u1[j]; }
#pragma unroll
    for (int nt = 0; nt < 8; ++nt) {
      v16h b = *(const v16h*)(pB + (((kt * 8 + nt) * 32 + lane) << 4));
      acc[nt] = WMMA_F16(a, b, acc[nt]);
    }
  }
  // bias + scatter-add
#pragma unroll
  for (int nt = 0; nt < 8; ++nt) {
    const int n = nt * 16 + m16;
    const float bv = bb[n];
#pragma unroll
    for (int r = 0; r < 8; ++r) {
      const int m = r + hi * 8;
      const int d = sDst[w][m];
      if (d >= 0) unsafeAtomicAdd(&hsum[(long)d * 128 + n], acc[nt][r] + bv);
    }
  }
}

// ---------------------------------------------------------------------------
// Finalize layer-1 nodes: h = relu(h_sum / max(cnt,1)), stored f16.
// ---------------------------------------------------------------------------
__global__ void node_h_kernel(const float* __restrict__ hsum, const float* __restrict__ cnt,
                              _Float16* __restrict__ h16, long total) {
  long i = (long)blockIdx.x * blockDim.x + threadIdx.x;
  if (i >= total) return;
  long node = i >> 7;
  float c = cnt[node];
  c = c > 1.f ? c : 1.f;
  float v = hsum[i] / c;
  h16[i] = (_Float16)(v > 0.f ? v : 0.f);
}

// ---------------------------------------------------------------------------
// Edge MLP layer 2: in = cat[h[dst], h[src]] (256 f16), relu(in@Wa+ba)@Wb+bb
// -> atomic scatter-add into g_sum[dst].
// ---------------------------------------------------------------------------
__global__ __launch_bounds__(256) void edge_mlp2_kernel(
    const _Float16* __restrict__ h16, const int* __restrict__ ei, long E,
    const _Float16* __restrict__ pA, const _Float16* __restrict__ pB,
    const float* __restrict__ ba, const float* __restrict__ bb,
    float* __restrict__ gsum) {
  __shared__ __align__(32) _Float16 sStage[8][16 * 64];
  __shared__ int sDst[8][16];
  const int w    = threadIdx.x >> 5;
  const int lane = threadIdx.x & 31;
  const int m16  = lane & 15;
  const int hi   = lane >> 4;
  const long base = ((long)blockIdx.x * 8 + w) * 16;
  long e = base + m16;
  const bool valid = (e < E);
  const long ec = valid ? e : 0;
  const int src = ei[ec];
  const int dst = ei[E + ec];
  if (hi == 0) sDst[w][m16] = valid ? dst : -1;
  const _Float16* rowD = h16 + (long)dst * 128;
  const _Float16* rowS = h16 + (long)src * 128;

  v8f acc[4];
#pragma unroll
  for (int nt = 0; nt < 4; ++nt) acc[nt] = vzero8();

  // GEMM1: [16x256] @ [256x64]
#pragma unroll
  for (int kt = 0; kt < 8; ++kt) {
    v16h a;
    const int f0 = kt * 32 + hi * 8;
    const int f1 = f0 + 16;
    const _Float16* p0 = (f0 < 128) ? (rowD + f0) : (rowS + (f0 - 128));
    const _Float16* p1 = (f1 < 128) ? (rowD + f1) : (rowS + (f1 - 128));
    v8h u0 = *(const v8h*)p0;
    v8h u1 = *(const v8h*)p1;
#pragma unroll
    for (int j = 0; j < 8; ++j) { a[j] = u0[j]; a[8 + j] = u1[j]; }
#pragma unroll
    for (int nt = 0; nt < 4; ++nt) {
      v16h b = *(const v16h*)(pA + (((kt * 4 + nt) * 32 + lane) << 4));
      acc[nt] = WMMA_F16(a, b, acc[nt]);
    }
  }
  _Float16* st = sStage[w];
#pragma unroll
  for (int nt = 0; nt < 4; ++nt) {
    const int n = nt * 16 + m16;
    const float bv = ba[n];
#pragma unroll
    for (int r = 0; r < 8; ++r) {
      const int m = r + hi * 8;
      float v = acc[nt][r] + bv;
      st[m * 64 + n] = (_Float16)(v > 0.f ? v : 0.f);
    }
  }
  // GEMM2: [16x64] @ [64x64]
#pragma unroll
  for (int nt = 0; nt < 4; ++nt) acc[nt] = vzero8();
#pragma unroll
  for (int kt = 0; kt < 2; ++kt) {
    v16h a;
    const int f0 = kt * 32 + hi * 8;
    v8h u0 = *(const v8h*)(st + m16 * 64 + f0);
    v8h u1 = *(const v8h*)(st + m16 * 64 + f0 + 16);
#pragma unroll
    for (int j = 0; j < 8; ++j) { a[j] = u0[j]; a[8 + j] = u1[j]; }
#pragma unroll
    for (int nt = 0; nt < 4; ++nt) {
      v16h b = *(const v16h*)(pB + (((kt * 4 + nt) * 32 + lane) << 4));
      acc[nt] = WMMA_F16(a, b, acc[nt]);
    }
  }
#pragma unroll
  for (int nt = 0; nt < 4; ++nt) {
    const int n = nt * 16 + m16;
    const float bv = bb[n];
#pragma unroll
    for (int r = 0; r < 8; ++r) {
      const int m = r + hi * 8;
      const int d = sDst[w][m];
      if (d >= 0) unsafeAtomicAdd(&gsum[(long)d * 64 + n], acc[nt][r] + bv);
    }
  }
}

// ---------------------------------------------------------------------------
// Node MLP head: per wave 16 nodes. x = cat[state(64), g(64), emb[mode](16)]
// padded to K=160; relu(x@w1+b1); relu(@w2+b2); mean = @wm+bm;
// log_std = clip(@ws+bs, -20, 2).
// ---------------------------------------------------------------------------
__global__ __launch_bounds__(128) void node_mlp_kernel(
    const float* __restrict__ state, const float* __restrict__ gsum,
    const float* __restrict__ cnt, const float* __restrict__ emb,
    const int* __restrict__ mode,
    const _Float16* __restrict__ pw1, const float* __restrict__ b1v,
    const _Float16* __restrict__ pw2, const float* __restrict__ b2v,
    const _Float16* __restrict__ pwm, const float* __restrict__ bmv,
    const _Float16* __restrict__ pws, const float* __restrict__ bsv,
    float* __restrict__ out_mean, float* __restrict__ out_ls, long N) {
  __shared__ __align__(32) _Float16 sA[4][16 * 256];
  __shared__ __align__(32) _Float16 sB[4][16 * 256];
  const int w    = threadIdx.x >> 5;
  const int lane = threadIdx.x & 31;
  const int m16  = lane & 15;
  const int hi   = lane >> 4;
  const long base = ((long)blockIdx.x * 4 + w) * 16;
  _Float16* A = sA[w];
  _Float16* B = sB[w];

  // build [16 x 160] input tile (cols 144..159 zero-padded), row stride 256
  for (int idx = lane; idx < 16 * 160; idx += 32) {
    int m = idx / 160, f = idx % 160;
    long node = base + m;
    if (node >= N) node = N - 1;
    float v;
    if (f < 64) v = state[node * 64 + f];
    else if (f < 128) {
      float c = cnt[node];
      c = c > 1.f ? c : 1.f;
      v = gsum[node * 64 + (f - 64)] / c;
    } else if (f < 144) v = emb[(long)mode[node] * 16 + (f - 128)];
    else v = 0.f;
    A[m * 256 + f] = (_Float16)v;
  }

  // layer 1: K=160 (padded 144) -> 256, two passes of 8 n-tiles
#pragma unroll
  for (int p = 0; p < 2; ++p) {
    v8f acc[8];
#pragma unroll
    for (int nt = 0; nt < 8; ++nt) acc[nt] = vzero8();
#pragma unroll
    for (int kt = 0; kt < 5; ++kt) {
      v16h a;
      const int f0 = kt * 32 + hi * 8;
      v8h u0 = *(const v8h*)(A + m16 * 256 + f0);
      v8h u1 = *(const v8h*)(A + m16 * 256 + f0 + 16);
#pragma unroll
      for (int j = 0; j < 8; ++j) { a[j] = u0[j]; a[8 + j] = u1[j]; }
#pragma unroll
      for (int nt = 0; nt < 8; ++nt) {
        v16h b = *(const v16h*)(pw1 + (((kt * 16 + p * 8 + nt) * 32 + lane) << 4));
        acc[nt] = WMMA_F16(a, b, acc[nt]);
      }
    }
#pragma unroll
    for (int nt = 0; nt < 8; ++nt) {
      const int n = (p * 8 + nt) * 16 + m16;
      const float bv = b1v[n];
#pragma unroll
      for (int r = 0; r < 8; ++r) {
        float v = acc[nt][r] + bv;
        B[(r + hi * 8) * 256 + n] = (_Float16)(v > 0.f ? v : 0.f);
      }
    }
  }

  // layer 2: 256 -> 256, B -> A
#pragma unroll
  for (int p = 0; p < 2; ++p) {
    v8f acc[8];
#pragma unroll
    for (int nt = 0; nt < 8; ++nt) acc[nt] = vzero8();
#pragma unroll
    for (int kt = 0; kt < 8; ++kt) {
      v16h a;
      const int f0 = kt * 32 + hi * 8;
      v8h u0 = *(const v8h*)(B + m16 * 256 + f0);
      v8h u1 = *(const v8h*)(B + m16 * 256 + f0 + 16);
#pragma unroll
      for (int j = 0; j < 8; ++j) { a[j] = u0[j]; a[8 + j] = u1[j]; }
#pragma unroll
      for (int nt = 0; nt < 8; ++nt) {
        v16h b = *(const v16h*)(pw2 + (((kt * 16 + p * 8 + nt) * 32 + lane) << 4));
        acc[nt] = WMMA_F16(a, b, acc[nt]);
      }
    }
#pragma unroll
    for (int nt = 0; nt < 8; ++nt) {
      const int n = (p * 8 + nt) * 16 + m16;
      const float bv = b2v[n];
#pragma unroll
      for (int r = 0; r < 8; ++r) {
        float v = acc[nt][r] + bv;
        A[(r + hi * 8) * 256 + n] = (_Float16)(v > 0.f ? v : 0.f);
      }
    }
  }

  // heads: 256 -> 8 (mean) and 256 -> 8 (log_std), N padded to 16
  v8f am = vzero8(), as = vzero8();
#pragma unroll
  for (int kt = 0; kt < 8; ++kt) {
    v16h a;
    const int f0 = kt * 32 + hi * 8;
    v8h u0 = *(const v8h*)(A + m16 * 256 + f0);
    v8h u1 = *(const v8h*)(A + m16 * 256 + f0 + 16);
#pragma unroll
    for (int j = 0; j < 8; ++j) { a[j] = u0[j]; a[8 + j] = u1[j]; }
    v16h bM = *(const v16h*)(pwm + ((kt * 32 + lane) << 4));
    v16h bS = *(const v16h*)(pws + ((kt * 32 + lane) << 4));
    am = WMMA_F16(a, bM, am);
    as = WMMA_F16(a, bS, as);
  }
  if (m16 < 8) {
    const float bm0 = bmv[m16];
    const float bs0 = bsv[m16];
#pragma unroll
    for (int r = 0; r < 8; ++r) {
      long node = base + r + hi * 8;
      if (node < N) {
        out_mean[node * 8 + m16] = am[r] + bm0;
        float ls = as[r] + bs0;
        ls = ls < -20.f ? -20.f : (ls > 2.f ? 2.f : ls);
        out_ls[node * 8 + m16] = ls;
      }
    }
  }
}

// ---------------------------------------------------------------------------
// Host launcher
// ---------------------------------------------------------------------------
extern "C" void kernel_launch(void* const* d_in, const int* in_sizes, int n_in,
                              void* d_out, int out_size, void* d_ws, size_t ws_size,
                              hipStream_t stream) {
  const float* state = (const float*)d_in[0];
  const float* xn    = (const float*)d_in[1];
  const int*   ei    = (const int*)d_in[2];
  const int*   mode  = (const int*)d_in[3];
  const float* w_g1a = (const float*)d_in[4];
  const float* b_g1a = (const float*)d_in[5];
  const float* w_g1b = (const float*)d_in[6];
  const float* b_g1b = (const float*)d_in[7];
  const float* w_g2a = (const float*)d_in[8];
  const float* b_g2a = (const float*)d_in[9];
  const float* w_g2b = (const float*)d_in[10];
  const float* b_g2b = (const float*)d_in[11];
  const float* emb   = (const float*)d_in[12];
  const float* w1    = (const float*)d_in[13];
  const float* b1    = (const float*)d_in[14];
  const float* w2    = (const float*)d_in[15];
  const float* b2    = (const float*)d_in[16];
  const float* wm    = (const float*)d_in[17];
  const float* bm    = (const float*)d_in[18];
  const float* wsw   = (const float*)d_in[19];
  const float* bs    = (const float*)d_in[20];

  const long N = in_sizes[0] / 64;
  const long E = in_sizes[2] / 2;

  char* wsb = (char*)d_ws;
  size_t off = 0;
  auto carve = [&](size_t bytes) -> void* {
    void* p = wsb + off;
    off += (bytes + 255) & ~(size_t)255;
    return p;
  };
  const size_t zbeg = off;
  float* h_sum = (float*)carve((size_t)N * 128 * sizeof(float));
  float* cnt   = (float*)carve((size_t)N * sizeof(float));
  float* g_sum = (float*)carve((size_t)N * 64 * sizeof(float));
  const size_t zend = off;
  _Float16* xn16 = (_Float16*)carve((size_t)N * 64 * sizeof(_Float16));
  _Float16* h16  = (_Float16*)carve((size_t)N * 128 * sizeof(_Float16));
  _Float16* pg1a = (_Float16*)carve((size_t)4 * 8 * 512 * 2);
  _Float16* pg1b = (_Float16*)carve((size_t)4 * 8 * 512 * 2);
  _Float16* pg2a = (_Float16*)carve((size_t)8 * 4 * 512 * 2);
  _Float16* pg2b = (_Float16*)carve((size_t)2 * 4 * 512 * 2);
  _Float16* pw1  = (_Float16*)carve((size_t)5 * 16 * 512 * 2);
  _Float16* pw2  = (_Float16*)carve((size_t)8 * 16 * 512 * 2);
  _Float16* pwm  = (_Float16*)carve((size_t)8 * 1 * 512 * 2);
  _Float16* pws  = (_Float16*)carve((size_t)8 * 1 * 512 * 2);

  // zero accumulators (sums + counts)
  long zcount = (long)((zend - zbeg) / 4);
  zero_kernel<<<dim3((unsigned)((zcount + 255) / 256)), dim3(256), 0, stream>>>(
      (float*)(wsb + zbeg), zcount);

  // pre-convert node features to f16
  long xcount = N * 64;
  to_f16_kernel<<<dim3((unsigned)((xcount + 255) / 256)), dim3(256), 0, stream>>>(
      xn, xn16, xcount);

  // pack all weights into WMMA B-fragment order (f16)
  auto pk = [&](const float* W, _Float16* dst, int K, int Nn, int KT, int NT) {
    int total = KT * NT * 512;
    pack_w_kernel<<<dim3((unsigned)((total + 255) / 256)), dim3(256), 0, stream>>>(
        W, dst, K, Nn, KT, NT);
  };
  pk(w_g1a, pg1a, 128, 128, 4, 8);
  pk(w_g1b, pg1b, 128, 128, 4, 8);
  pk(w_g2a, pg2a, 256, 64, 8, 4);
  pk(w_g2b, pg2b, 64, 64, 2, 4);
  pk(w1, pw1, 144, 256, 5, 16);
  pk(w2, pw2, 256, 256, 8, 16);
  pk(wm, pwm, 256, 8, 8, 1);
  pk(wsw, pws, 256, 8, 8, 1);

  const unsigned eblocks = (unsigned)((E + 127) / 128);  // 8 waves x 16 edges
  edge_mlp1_kernel<<<dim3(eblocks), dim3(256), 0, stream>>>(
      xn16, ei, E, pg1a, pg1b, b_g1a, b_g1b, h_sum, cnt);

  long t1 = N * 128;
  node_h_kernel<<<dim3((unsigned)((t1 + 255) / 256)), dim3(256), 0, stream>>>(
      h_sum, cnt, h16, t1);

  edge_mlp2_kernel<<<dim3(eblocks), dim3(256), 0, stream>>>(
      h16, ei, E, pg2a, pg2b, b_g2a, b_g2b, g_sum);

  float* out_mean = (float*)d_out;
  float* out_ls   = out_mean + (size_t)N * 8;
  const unsigned nblocks = (unsigned)((N + 63) / 64);  // 4 waves x 16 nodes
  node_mlp_kernel<<<dim3(nblocks), dim3(128), 0, stream>>>(
      state, g_sum, cnt, emb, mode, pw1, b1, pw2, b2, pwm, bm, pws, bs,
      out_mean, out_ls, N);

  (void)n_in; (void)out_size; (void)ws_size;
}
